// SO2_Convolution_7945689498243
// MI455X (gfx1250) — matrix operand order repaired and checked by reading, not compiled, not run on hardware
//
#include <hip/hip_runtime.h>
#include <hip/hip_bf16.h>
#include <math.h>

// ---------------------------------------------------------------------------
// SO2 convolution, fused, bf16 WMMA (gfx1250 / CDNA5, wave32)
// LMAX=6 MMAX=2 C=128 MOUT=128  NCOEF=29  MSIZE={7,6,5}
// ---------------------------------------------------------------------------

typedef __bf16 bf16_t;
typedef bf16_t v16bf __attribute__((ext_vector_type(16)));
typedef float  v8f   __attribute__((ext_vector_type(8)));
typedef float  fv4   __attribute__((ext_vector_type(4)));

#define ROWSTRIDE 3712   // 29 * 128
#define TILE_E    16

// permutation new->old from _build_to_m(): m=0 block, m=+1, m=-1, m=+2, m=-2
__constant__ int kOrder[29] = {
    0, 2, 6, 11, 16, 21, 26,          // m=0   (rows 0..6)
    3, 7, 12, 17, 22, 27,             // m=+1  (rows 7..12, real)
    1, 5, 10, 15, 20, 25,             // m=-1  (rows 13..18, imag)
    8, 13, 18, 23, 28,                // m=+2  (rows 19..23, real)
    4, 9, 14, 19, 24                  // m=-2  (rows 24..28, imag)
};

// bf16 weight regions inside d_ws (element offsets)
#define OFF_WR0 0u
#define OFF_WR1 32768u        // 256*128
#define OFF_WR2 98304u        // +256*256
#define OFF_W0  688128u       // +2304*256
#define OFF_W1  1490944u      // +896*896
#define OFF_W2  2670592u      // +1536*768
#define W_TOTAL 3489792u      // +1280*640

union Frag { fv4 f[2]; v16bf v; };

// A fragment: 16x32 bf16 tile, row-major [16][kstride] in LDS.
// lane L: row = L&15, half h = L>>4.
//   VGPR0-3 : K = k0 + 8h   .. +7   (16 contiguous bytes)
//   VGPR4-7 : K = k0 +16+8h .. +7   (16 contiguous bytes)
__device__ __forceinline__ v16bf ldsA(const bf16_t* base, int kstride, int k0, int lane) {
    const int m = lane & 15, h = lane >> 4;
    const bf16_t* p = base + m * kstride + k0 + 8 * h;
    Frag u;
    u.f[0] = *(const fv4*)(p);
    u.f[1] = *(const fv4*)(p + 16);
    return u.v;
}

// B fragment: 32x16 bf16 tile of B[k][n] = W[n][k], W row-major (N,K) in global.
// lane L: col n = n0 + (L&15), half h = L>>4; K = k0+8h..+7 and k0+16+8h..+7
// -> two 16-byte chunks per lane.
__device__ __forceinline__ v16bf glbB(const bf16_t* __restrict__ W, int K,
                                      int n0, int k0, int lane) {
    const int n = n0 + (lane & 15), h = lane >> 4;
    const bf16_t* p = W + (size_t)n * K + k0 + 8 * h;
    Frag u;
    u.f[0] = *(const fv4*)(p);
    u.f[1] = *(const fv4*)(p + 16);
    return u.v;
}

__device__ __forceinline__ v8f wmma_bf16(v16bf a, v16bf b, v8f c) {
    return __builtin_amdgcn_wmma_f32_16x16x32_bf16(
        /*neg_a=*/false, a, /*neg_b=*/false, b,
        /*c_mod=*/(short)0, c, /*reuse_a=*/false, /*reuse_b=*/false);
}

// silu via hardware v_rcp_f32 (result feeds a bf16 quantization; 1-ulp rcp is
// far below the bf16 rounding error, and avoids the IEEE div_scale/div_fmas
// expansion that dominated the VALU mix otherwise).
__device__ __forceinline__ float silu(float x) {
    return x * __builtin_amdgcn_rcpf(1.0f + __expf(-x));
}

// ---------------------------------------------------------------------------
// Kernel 1: convert all six weight matrices f32 -> bf16 into workspace
// ---------------------------------------------------------------------------
__global__ void convert_weights(const float* __restrict__ Wr0,
                                const float* __restrict__ Wr1,
                                const float* __restrict__ Wr2,
                                const float* __restrict__ W0,
                                const float* __restrict__ W1,
                                const float* __restrict__ W2,
                                bf16_t* __restrict__ ws) {
    size_t i = (size_t)blockIdx.x * blockDim.x + threadIdx.x;
    if (i >= W_TOTAL) return;
    float v;
    if      (i < OFF_WR1) v = Wr0[i - OFF_WR0];
    else if (i < OFF_WR2) v = Wr1[i - OFF_WR1];
    else if (i < OFF_W0)  v = Wr2[i - OFF_WR2];
    else if (i < OFF_W1)  v = W0[i - OFF_W0];
    else if (i < OFF_W2)  v = W1[i - OFF_W1];
    else                  v = W2[i - OFF_W2];
    ws[i] = (bf16_t)v;
}

// ---------------------------------------------------------------------------
// Kernel 2: fused SO2 convolution. One block = 16 edges, 8 wave32s.
// ---------------------------------------------------------------------------
__global__ __launch_bounds__(256)
void so2_fused(const float* __restrict__ x,        // (n,29,128)
               const float* __restrict__ xe,       // (n,128)
               const float* __restrict__ br0,      // (256)
               const float* __restrict__ br1,      // (256)
               const float* __restrict__ br2,      // (2304)
               const float* __restrict__ b0,       // (896)
               const bf16_t* __restrict__ ws,
               float* __restrict__ out,            // (n,29,128)
               int n_edges) {
    __shared__ alignas(16) bf16_t sXe[TILE_E * 128];   //  4 KB
    __shared__ alignas(16) bf16_t sH0[TILE_E * 256];   //  8 KB
    __shared__ alignas(16) bf16_t sH1[TILE_E * 256];   //  8 KB
    __shared__ alignas(16) bf16_t sA0[TILE_E * 896];   // 28 KB  m=0 input
    __shared__ alignas(16) bf16_t sA1[2 * TILE_E * 768]; // 48 KB m=1 (rows 2e/2e+1)
    __shared__ alignas(16) bf16_t sA2[2 * TILE_E * 640]; // 40 KB m=2

    const bf16_t* Wr0b = ws + OFF_WR0;
    const bf16_t* Wr1b = ws + OFF_WR1;
    const bf16_t* Wr2b = ws + OFF_WR2;
    const bf16_t* W0b  = ws + OFF_W0;
    const bf16_t* W1b  = ws + OFF_W1;
    const bf16_t* W2b  = ws + OFF_W2;

    const int tid  = threadIdx.x;
    const int wave = tid >> 5;
    const int lane = tid & 31;
    const int nl   = lane & 15;
    const int hh   = lane >> 4;
    const int e0   = blockIdx.x * TILE_E;

    // ---- Stage 0: x_edge tile -> LDS (bf16) ------------------------------
    for (int i = tid; i < TILE_E * 128; i += 256) {
        const int e = i >> 7, c = i & 127;
        const int eg = min(e0 + e, n_edges - 1);
        sXe[i] = (bf16_t)xe[(size_t)eg * 128 + c];
    }
    __syncthreads();

    // ---- Layer 1: h0 = silu(xe @ Wr0^T + br0)  M=16 N=256 K=128 ----------
    for (int nt = wave; nt < 16; nt += 8) {
        v8f acc = {};
#pragma unroll
        for (int k0 = 0; k0 < 128; k0 += 32)
            acc = wmma_bf16(ldsA(sXe, 128, k0, lane),
                            glbB(Wr0b, 128, nt * 16, k0, lane), acc);
        const int n = nt * 16 + nl;
        const float bias = br0[n];
#pragma unroll
        for (int v = 0; v < 8; ++v)
            sH0[(v + 8 * hh) * 256 + n] = (bf16_t)silu(acc[v] + bias);
    }
    __syncthreads();

    // ---- Layer 2: h1 = silu(h0 @ Wr1^T + br1)  M=16 N=256 K=256 ----------
    for (int nt = wave; nt < 16; nt += 8) {
        v8f acc = {};
#pragma unroll
        for (int k0 = 0; k0 < 256; k0 += 32)
            acc = wmma_bf16(ldsA(sH0, 256, k0, lane),
                            glbB(Wr1b, 256, nt * 16, k0, lane), acc);
        const int n = nt * 16 + nl;
        const float bias = br1[n];
#pragma unroll
        for (int v = 0; v < 8; ++v)
            sH1[(v + 8 * hh) * 256 + n] = (bf16_t)silu(acc[v] + bias);
    }
    __syncthreads();

    // ---- Layer 3: rad = h1 @ Wr2^T + br2, fused with xp*rad gather -------
    // rad cols: [0,896) -> m=0 ; [896,1664) -> m=1 ; [1664,2304) -> m=2
    for (int nt = wave; nt < 144; nt += 8) {
        v8f acc = {};
#pragma unroll
        for (int k0 = 0; k0 < 256; k0 += 32)
            acc = wmma_bf16(ldsA(sH1, 256, k0, lane),
                            glbB(Wr2b, 256, nt * 16, k0, lane), acc);
        const int j = nt * 16 + nl;
        const float bias = br2[j];
        if (j < 896) {
            const int coef = j >> 7, chan = j & 127;
            const int old = kOrder[coef];
#pragma unroll
            for (int v = 0; v < 8; ++v) {
                const int e = v + 8 * hh;
                const int eg = min(e0 + e, n_edges - 1);
                const float rv = acc[v] + bias;
                const float xv = x[(size_t)eg * ROWSTRIDE + old * 128 + chan];
                sA0[e * 896 + j] = (bf16_t)(xv * rv);
            }
        } else if (j < 1664) {
            const int jj = j - 896, coef = jj >> 7, chan = jj & 127;
            const int oldR = kOrder[7 + coef], oldI = kOrder[13 + coef];
#pragma unroll
            for (int v = 0; v < 8; ++v) {
                const int e = v + 8 * hh;
                const int eg = min(e0 + e, n_edges - 1);
                const float rv = acc[v] + bias;
                const size_t xb = (size_t)eg * ROWSTRIDE + chan;
                sA1[(2 * e) * 768 + jj]     = (bf16_t)(x[xb + oldR * 128] * rv);
                sA1[(2 * e + 1) * 768 + jj] = (bf16_t)(x[xb + oldI * 128] * rv);
            }
        } else {
            const int jj = j - 1664, coef = jj >> 7, chan = jj & 127;
            const int oldR = kOrder[19 + coef], oldI = kOrder[24 + coef];
#pragma unroll
            for (int v = 0; v < 8; ++v) {
                const int e = v + 8 * hh;
                const int eg = min(e0 + e, n_edges - 1);
                const float rv = acc[v] + bias;
                const size_t xb = (size_t)eg * ROWSTRIDE + chan;
                sA2[(2 * e) * 640 + jj]     = (bf16_t)(x[xb + oldR * 128] * rv);
                sA2[(2 * e + 1) * 640 + jj] = (bf16_t)(x[xb + oldI * 128] * rv);
            }
        }
    }
    __syncthreads();

    // ---- Output GEMMs ----------------------------------------------------
    // work items: 56 (m=0 N-tiles) + 96 (m=1: 2 M-tiles x 48 N-pair-tiles)
    //                              + 80 (m=2: 2 M-tiles x 40 N-pair-tiles)
    for (int w = wave; w < 232; w += 8) {
        if (w < 56) {
            // m=0 : (16 x 896) @ (896 x 896) + b0, scatter rows 0..6
            const int nt = w;
            v8f acc = {};
            for (int k0 = 0; k0 < 896; k0 += 32)
                acc = wmma_bf16(ldsA(sA0, 896, k0, lane),
                                glbB(W0b, 896, nt * 16, k0, lane), acc);
            const int n = nt * 16 + nl;
            const float bias = b0[n];
            const int coef = n >> 7, chan = n & 127;
            const int old = kOrder[coef];
#pragma unroll
            for (int v = 0; v < 8; ++v) {
                const int e = e0 + v + 8 * hh;
                if (e < n_edges)
                    out[(size_t)e * ROWSTRIDE + old * 128 + chan] = acc[v] + bias;
            }
        } else if (w < 152) {
            // m=1 : (32 x 768) @ (768 x 1536); pair tiles (t, t+48)
            const int q = w - 56, mt = q & 1, t = q >> 1;
            v8f accA = {}, accB = {};
            const bf16_t* Abase = sA1 + mt * 16 * 768;
            for (int k0 = 0; k0 < 768; k0 += 32) {
                const v16bf a = ldsA(Abase, 768, k0, lane);
                accA = wmma_bf16(a, glbB(W1b, 768, t * 16, k0, lane), accA);
                accB = wmma_bf16(a, glbB(W1b, 768, (48 + t) * 16, k0, lane), accB);
            }
            const int jj = t * 16 + nl, coef = jj >> 7, chan = jj & 127;
            const int oldR = kOrder[7 + coef], oldI = kOrder[13 + coef];
#pragma unroll
            for (int v = 0; v < 8; v += 2) {
                const int row = mt * 16 + v + 8 * hh;   // even = real input row
                const int e = e0 + (row >> 1);
                if (e < n_edges) {
                    const size_t base = (size_t)e * ROWSTRIDE + chan;
                    out[base + oldR * 128] = accA[v]     - accB[v + 1];
                    out[base + oldI * 128] = accA[v + 1] + accB[v];
                }
            }
        } else {
            // m=2 : (32 x 640) @ (640 x 1280); pair tiles (t, t+40)
            const int q = w - 152, mt = q & 1, t = q >> 1;
            v8f accA = {}, accB = {};
            const bf16_t* Abase = sA2 + mt * 16 * 640;
            for (int k0 = 0; k0 < 640; k0 += 32) {
                const v16bf a = ldsA(Abase, 640, k0, lane);
                accA = wmma_bf16(a, glbB(W2b, 640, t * 16, k0, lane), accA);
                accB = wmma_bf16(a, glbB(W2b, 640, (40 + t) * 16, k0, lane), accB);
            }
            const int jj = t * 16 + nl, coef = jj >> 7, chan = jj & 127;
            const int oldR = kOrder[19 + coef], oldI = kOrder[24 + coef];
#pragma unroll
            for (int v = 0; v < 8; v += 2) {
                const int row = mt * 16 + v + 8 * hh;
                const int e = e0 + (row >> 1);
                if (e < n_edges) {
                    const size_t base = (size_t)e * ROWSTRIDE + chan;
                    out[base + oldR * 128] = accA[v]     - accB[v + 1];
                    out[base + oldI * 128] = accA[v + 1] + accB[v];
                }
            }
        }
    }
}

// ---------------------------------------------------------------------------
extern "C" void kernel_launch(void* const* d_in, const int* in_sizes, int n_in,
                              void* d_out, int out_size, void* d_ws, size_t ws_size,
                              hipStream_t stream) {
    const float* x      = (const float*)d_in[0];
    const float* x_edge = (const float*)d_in[1];
    // d_in[2] = to_m (fixed structural permutation; hard-coded as kOrder)
    const float* Wr0 = (const float*)d_in[3];
    const float* br0 = (const float*)d_in[4];
    const float* Wr1 = (const float*)d_in[5];
    const float* br1 = (const float*)d_in[6];
    const float* Wr2 = (const float*)d_in[7];
    const float* br2 = (const float*)d_in[8];
    const float* W0  = (const float*)d_in[9];
    const float* b0  = (const float*)d_in[10];
    const float* W1  = (const float*)d_in[11];
    const float* W2  = (const float*)d_in[12];

    bf16_t* wsb = (bf16_t*)d_ws;
    float*  out = (float*)d_out;

    const int n_edges = in_sizes[0] / ROWSTRIDE;

    const int cvt_blocks = (int)((W_TOTAL + 255u) / 256u);
    convert_weights<<<cvt_blocks, 256, 0, stream>>>(Wr0, Wr1, Wr2, W0, W1, W2, wsb);

    const int grid = (n_edges + TILE_E - 1) / TILE_E;
    so2_fused<<<grid, 256, 0, stream>>>(x, x_edge, br0, br1, br2, b0, wsb, out, n_edges);
}
